// MultiMambaBlockJIT_5832565588116
// MI455X (gfx1250) — matrix operand order repaired and checked by reading, not compiled, not run on hardware
//
#include <hip/hip_runtime.h>
#include <hip/hip_bf16.h>
#include <math.h>

// ---------------- model constants ----------------
#define D_MODEL   512
#define NUM_N     4
#define D_STATE   16
#define D_CONV    4
#define D_INNER   1024
#define DT_RANK   32
#define B_SZ      2
#define L_SEQ     1024
#define M_ROWS    (B_SZ * NUM_N * L_SEQ)      // 8192
#define E2        (2 * D_INNER)               // 2048
#define P_COLS    (DT_RANK + 2 * D_STATE)     // 64

typedef __attribute__((ext_vector_type(16))) _Float16 v16h;
typedef __attribute__((ext_vector_type(8)))  float    v8f;

// LDS byte offset of a __shared__ object: generic flat address of LDS is
// {SHARED_BASE aperture (hi 32), LDS offset (lo 32)} -> truncate.
__device__ __forceinline__ unsigned lds_off(const void* p) {
    return (unsigned)(unsigned long long)(size_t)p;
}

__device__ __forceinline__ void async_b128(unsigned lds, const void* gaddr) {
    unsigned long long ga = (unsigned long long)(size_t)gaddr;
    asm volatile("global_load_async_to_lds_b128 %0, %1, off"
                 :: "v"(lds), "v"(ga) : "memory");
}

// ======================================================================
// WMMA GEMM:  C[m,n] = sum_k A[m, k] * W[n, k]   (W row-major (Nout, K))
// Block: 256 threads = 8 waves. Block tile 64(M) x 64(N).
// K tiled by 32.  f32 tiles DMA'd memory->LDS with
// GLOBAL_LOAD_ASYNC_TO_LDS_B128 into double buffers (ASYNCcnt pipelined),
// converted to f16 at fragment build, f32 accumulate in v_wmma.
// EPI: 0 = plain store, 1 = min(softplus(v + bias[n]), 2.5)
// ======================================================================
template <int EPI>
__global__ __launch_bounds__(256)
void wmma_gemm_kernel(const float* __restrict__ A,
                      const float* __restrict__ W,
                      float* __restrict__ C,
                      int M, int K, int Nout, int lda,
                      const float* __restrict__ bias)
{
    __shared__ alignas(16) float Af[2][64][36];   // [buf][m][k] stride 144B
    __shared__ alignas(16) float Bf[2][64][36];   // [buf][n][k] (W tile)

    const int tid  = threadIdx.x;
    const int lane = tid & 31;
    const int wave = tid >> 5;      // 0..7
    const int wm   = wave >> 1;     // 0..3
    const int wn   = wave & 1;      // 0..1

    const int n0 = blockIdx.x * 64;
    const int m0 = blockIdx.y * 64;

    // staging coords: 512 16B-chunks per 64x32 tile -> 2 per thread per tile
    const int r0 = tid >> 3;          // rows 0..31
    const int r1 = r0 + 32;           // rows 32..63
    const int c4 = (tid & 7) * 4;     // f32 col 0,4,...,28 (16B aligned)

    v8f acc0 = {}; v8f acc1 = {};

    // issue the 4 async DMAs (2 A-chunks + 2 B-chunks) for one K-step
    auto issue = [&](int buf, int kk) {
        async_b128(lds_off(&Af[buf][r0][c4]), A + (size_t)(m0 + r0) * lda + kk + c4);
        async_b128(lds_off(&Af[buf][r1][c4]), A + (size_t)(m0 + r1) * lda + kk + c4);
        async_b128(lds_off(&Bf[buf][r0][c4]), W + (size_t)(n0 + r0) * K   + kk + c4);
        async_b128(lds_off(&Bf[buf][r1][c4]), W + (size_t)(n0 + r1) * K   + kk + c4);
    };

    int buf = 0;
    issue(0, 0);

    for (int kk = 0; kk < K; kk += 32) {
        if (kk + 32 < K) {
            issue(buf ^ 1, kk + 32);
            // oldest 4 (current buffer) complete; next 4 stay in flight
            asm volatile("s_wait_asynccnt 0x4" ::: "memory");
        } else {
            asm volatile("s_wait_asynccnt 0x0" ::: "memory");
        }
        __syncthreads();

        // A fragment: contiguous K order, same as B (lanes<16: K 0-15; lanes>=16: K 16-31)
        const int mrow = wm * 16 + (lane & 15);
        const int kbA  = (lane < 16) ? 0 : 16;
        v16h a;
        #pragma unroll
        for (int j = 0; j < 8; ++j) {
            a[j]     = (_Float16)Af[buf][mrow][kbA + j];
            a[j + 8] = (_Float16)Af[buf][mrow][kbA + 8 + j];
        }
        // B fragments: contiguous K order, same as A (lanes<16: K 0-15; lanes>=16: K 16-31)
        const int kbB = (lane < 16) ? 0 : 16;
        const int nc0 = wn * 32 + (lane & 15);
        v16h b0, b1;
        #pragma unroll
        for (int j = 0; j < 16; ++j) {
            b0[j] = (_Float16)Bf[buf][nc0][kbB + j];
            b1[j] = (_Float16)Bf[buf][nc0 + 16][kbB + j];
        }

        acc0 = __builtin_amdgcn_wmma_f32_16x16x32_f16(false, a, false, b0,
                                                      (short)0, acc0, false, false);
        acc1 = __builtin_amdgcn_wmma_f32_16x16x32_f16(false, a, false, b1,
                                                      (short)0, acc1, false, false);
        __syncthreads();
        buf ^= 1;
    }

    // store: C/D layout: VGPR r -> lanes 0-15 (M=r), lanes 16-31 (M=r+8)
    const int rbase = m0 + wm * 16 + ((lane >= 16) ? 8 : 0);
    const int col0  = n0 + wn * 32 + (lane & 15);
    #pragma unroll
    for (int r = 0; r < 8; ++r) {
        const long row = rbase + r;
        float v0 = acc0[r], v1 = acc1[r];
        if (EPI == 1) {
            float t0 = v0 + bias[col0];
            float t1 = v1 + bias[col0 + 16];
            t0 = (t0 > 20.f) ? t0 : logf(1.f + expf(t0));
            t1 = (t1 > 20.f) ? t1 : logf(1.f + expf(t1));
            v0 = fminf(t0, 2.5f);
            v1 = fminf(t1, 2.5f);
        }
        C[row * Nout + col0]      = v0;
        C[row * Nout + col0 + 16] = v1;
    }
}

// ======================================================================
// Depthwise causal conv (K=4) + bias + SiLU.
// ======================================================================
__global__ __launch_bounds__(256)
void conv_silu_kernel(const float* __restrict__ XR,
                      const float* __restrict__ cw,
                      const float* __restrict__ cb,
                      float* __restrict__ XC)
{
    const int idx = blockIdx.x * 256 + threadIdx.x;   // m*1024 + d
    const int d = idx & (D_INNER - 1);
    const int m = idx >> 10;
    const int l = m & (L_SEQ - 1);
    const int n = (m >> 10) & (NUM_N - 1);
    const int c = (n << 10) + d;

    float s = cb[c];
    #pragma unroll
    for (int j = 0; j < D_CONV; ++j) {
        const int li = l - (D_CONV - 1) + j;
        if (li >= 0)
            s += cw[c * D_CONV + j] * XR[(long)(m - (D_CONV - 1) + j) * E2 + d];
    }
    XC[idx] = s / (1.f + expf(-s));   // silu
}

// ======================================================================
// SSM scan: one block per (sequence, 256-channel chunk). One thread per
// channel d; h[16] in registers; B_t/C_t broadcast async DMA'd to LDS.
// Writes z = (y + x*D) * silu(res) in-place over XC.
// ======================================================================
__global__ __launch_bounds__(256)
void ssm_scan_kernel(const float* __restrict__ XR,
                     float* __restrict__ XC,
                     const float* __restrict__ PROJ,
                     const float* __restrict__ DT,
                     const float* __restrict__ A_log,
                     const float* __restrict__ D_param)
{
    __shared__ alignas(16) float bc[2 * D_STATE];
    const int bn = blockIdx.x >> 2;
    const int d  = ((blockIdx.x & 3) << 8) + threadIdx.x;

    float Av[D_STATE];
    #pragma unroll
    for (int s = 0; s < D_STATE; ++s) Av[s] = -expf(A_log[d * D_STATE + s]);
    const float Dp = D_param[d];

    float h[D_STATE];
    #pragma unroll
    for (int s = 0; s < D_STATE; ++s) h[s] = 0.f;

    for (int l = 0; l < L_SEQ; ++l) {
        const long m = (long)(bn << 10) + l;
        if (threadIdx.x < 8)   // 8 lanes x 16B = 32 floats (B_t | C_t)
            async_b128(lds_off(&bc[threadIdx.x * 4]),
                       PROJ + m * P_COLS + DT_RANK + threadIdx.x * 4);
        asm volatile("s_wait_asynccnt 0x0" ::: "memory");
        __syncthreads();

        const float x  = XC[m * D_INNER + d];
        const float dt = DT[m * D_INNER + d];
        float y = 0.f;
        #pragma unroll
        for (int s = 0; s < D_STATE; ++s) {
            const float dA = __expf(dt * Av[s]);
            h[s] = dA * h[s] + (dt * bc[s]) * x;
            y += h[s] * bc[D_STATE + s];
        }
        y += x * Dp;

        const float res = XR[m * E2 + D_INNER + d];
        const float sil = res / (1.f + expf(-res));
        XC[m * D_INNER + d] = y * sil;
        __syncthreads();
    }
}

// ======================================================================
extern "C" void kernel_launch(void* const* d_in, const int* in_sizes, int n_in,
                              void* d_out, int out_size, void* d_ws, size_t ws_size,
                              hipStream_t stream)
{
    const float* x          = (const float*)d_in[0];   // (2,4,1024,512)
    const float* in_proj_w  = (const float*)d_in[1];   // (2048,512)
    const float* conv_w     = (const float*)d_in[2];   // (4096,1,4)
    const float* conv_b     = (const float*)d_in[3];   // (4096,)
    const float* x_proj_w   = (const float*)d_in[4];   // (64,1024)
    const float* dt_proj_w  = (const float*)d_in[5];   // (1024,32)
    const float* dt_proj_b  = (const float*)d_in[6];   // (1024,)
    const float* A_log      = (const float*)d_in[7];   // (1024,16)
    const float* D_param    = (const float*)d_in[8];   // (1024,)
    const float* out_proj_w = (const float*)d_in[9];   // (512,1024)
    float* out = (float*)d_out;                        // (2,4,1024,512)

    float* XR   = (float*)d_ws;                        // 8192*2048
    float* XC   = XR + (long)M_ROWS * E2;              // 8192*1024 (x_conv, then z)
    float* PROJ = XC + (long)M_ROWS * D_INNER;         // 8192*64
    float* DTB  = PROJ + (long)M_ROWS * P_COLS;        // 8192*1024

    // 1) in_proj: (8192x512) @ (512x2048) -> XR
    wmma_gemm_kernel<0><<<dim3(E2 / 64, M_ROWS / 64), 256, 0, stream>>>(
        x, in_proj_w, XR, M_ROWS, D_MODEL, E2, D_MODEL, nullptr);

    // 2) depthwise conv + silu -> XC
    conv_silu_kernel<<<(M_ROWS * D_INNER) / 256, 256, 0, stream>>>(
        XR, conv_w, conv_b, XC);

    // 3) x_proj: (8192x1024) @ (1024x64) -> PROJ
    wmma_gemm_kernel<0><<<dim3(P_COLS / 64, M_ROWS / 64), 256, 0, stream>>>(
        XC, x_proj_w, PROJ, M_ROWS, D_INNER, P_COLS, D_INNER, nullptr);

    // 4) dt_proj + bias + softplus + clamp: (8192x32) @ (32x1024) -> DTB
    wmma_gemm_kernel<1><<<dim3(D_INNER / 64, M_ROWS / 64), 256, 0, stream>>>(
        PROJ, dt_proj_w, DTB, M_ROWS, DT_RANK, D_INNER, P_COLS, dt_proj_b);

    // 5) selective scan + gating, z written in-place over XC
    ssm_scan_kernel<<<B_SZ * NUM_N * (D_INNER / 256), 256, 0, stream>>>(
        XR, XC, PROJ, DTB, A_log, D_param);

    // 6) out_proj: (8192x1024) @ (1024x512) -> out
    wmma_gemm_kernel<0><<<dim3(D_MODEL / 64, M_ROWS / 64), 256, 0, stream>>>(
        XC, out_proj_w, out, M_ROWS, D_INNER, D_MODEL, D_INNER, nullptr);
}
